// GatedDeltaNet_70798240907758
// MI455X (gfx1250) — compile-verified, matrix operand order checked by reading, not run on hardware
//
#include <hip/hip_runtime.h>
#include <hip/hip_bf16.h>

typedef __bf16 v16bf __attribute__((ext_vector_type(16)));
typedef __bf16 v8bf  __attribute__((ext_vector_type(8)));
typedef __bf16 v4bf  __attribute__((ext_vector_type(4)));
typedef float  v8f   __attribute__((ext_vector_type(8)));

static __device__ __forceinline__ __bf16 f2bf(float f) {
  unsigned u = __builtin_bit_cast(unsigned, f);
  u += 0x7FFFu + ((u >> 16) & 1u);               // round-to-nearest-even
  unsigned short h = (unsigned short)(u >> 16);
  return __builtin_bit_cast(__bf16, h);
}
static __device__ __forceinline__ float bf2f(__bf16 b) {
  unsigned short h = __builtin_bit_cast(unsigned short, b);
  unsigned u = ((unsigned)h) << 16;
  return __builtin_bit_cast(float, u);
}

// ISA 7.12.2: lane L's 16 fragment elements are two contiguous 8-elem runs:
//   e=0..7  -> K = k0 + (L/16)*8 + e
//   e=8..15 -> K = k0 + 16 + (L/16)*8 + (e-8)
// With the contraction index row-contiguous in LDS this is 2x ds_load_b128.
static __device__ __forceinline__ v16bf frag_ld(const __bf16* row, int k0, int lane) {
  const int hs = (lane >> 4) << 3;
  const v8bf lo = *reinterpret_cast<const v8bf*>(row + k0 + hs);
  const v8bf hi = *reinterpret_cast<const v8bf*>(row + k0 + 16 + hs);
  return __builtin_shufflevector(lo, hi, 0, 1, 2, 3, 4, 5, 6, 7,
                                 8, 9, 10, 11, 12, 13, 14, 15);
}
#define WMMA_BF16(a, b, c) \
  __builtin_amdgcn_wmma_f32_16x16x32_bf16(false, (a), false, (b), (short)0, (c), false, false)

// gfx1250 async copy: per-lane 16B global->LDS, tracked by ASYNCcnt.
static __device__ __forceinline__ void async_b128(unsigned lds_addr, unsigned long long gaddr) {
  asm volatile("global_load_async_to_lds_b128 %0, %1, off"
               :: "v"(lds_addr), "v"(gaddr) : "memory");
}
static __device__ __forceinline__ void wait_async0() {
  asm volatile("s_wait_asynccnt 0" ::: "memory");
}

// ---------------------------------------------------------------------------
// Kernel 1: fused [x @ w_qkv | x @ w_gate] with activation epilogue.
// M=2048, K=1024, N=4096 (3072 qkv + 1024 gate). 64x64 tile / block, 16 waves.
// ---------------------------------------------------------------------------
__global__ __launch_bounds__(512) void k_qkv_gate(
    const float* __restrict__ x, const float* __restrict__ w_qkv,
    const float* __restrict__ w_gate, __bf16* __restrict__ qb,
    __bf16* __restrict__ kb, __bf16* __restrict__ vb,
    float* __restrict__ gate) {
  __shared__ alignas(16) __bf16 As[64][32];   // (m, k) row-major
  __shared__ alignas(16) __bf16 Bt[64][32];   // (n, k): transposed for B-frag
  const int tid  = threadIdx.x;
  const int lane = tid & 31;
  const int wave = tid >> 5;
  const int tm   = wave >> 2;
  const int tn   = wave & 3;
  const int m0 = (int)(blockIdx.x >> 6) * 64;
  const int n0 = (int)(blockIdx.x & 63) * 64;
  const bool isGate = (n0 >= 3072);
  const float* wsrc = isGate ? w_gate : w_qkv;
  const int wld = isGate ? 1024 : 3072;
  const int nb  = isGate ? (n0 - 3072) : n0;

  const int arow = tid >> 3, ac = (tid & 7) << 2;    // A stage coords
  const int brow = tid >> 4, bc = (tid & 15) << 2;   // B stage coords

  v8f acc = {};
  for (int k0 = 0; k0 < 1024; k0 += 32) {
    {  // stage A: 64x32 f32 -> bf16 (one float4 per thread)
      float4 v = *reinterpret_cast<const float4*>(x + (size_t)(m0 + arow) * 1024 + k0 + ac);
      *reinterpret_cast<v4bf*>(&As[arow][ac]) = v4bf{f2bf(v.x), f2bf(v.y), f2bf(v.z), f2bf(v.w)};
    }
    {  // stage B transposed: read (k, n..n+3) coalesced, scatter to Bt[n][k]
      float4 v = *reinterpret_cast<const float4*>(wsrc + (size_t)(k0 + brow) * wld + nb + bc);
      Bt[bc + 0][brow] = f2bf(v.x); Bt[bc + 1][brow] = f2bf(v.y);
      Bt[bc + 2][brow] = f2bf(v.z); Bt[bc + 3][brow] = f2bf(v.w);
    }
    if (k0 + 32 < 1024) {  // prefetch next K tile (global_prefetch_b8)
      __builtin_prefetch(x + (size_t)(m0 + arow) * 1024 + k0 + 32 + ac, 0, 0);
      __builtin_prefetch(wsrc + (size_t)(k0 + 32 + brow) * wld + nb + bc, 0, 0);
    }
    __syncthreads();
    v16bf af = frag_ld(&As[tm * 16 + (lane & 15)][0], 0, lane);
    v16bf bf = frag_ld(&Bt[tn * 16 + (lane & 15)][0], 0, lane);
    acc = WMMA_BF16(af, bf, acc);
    __syncthreads();
  }

  const int ncol = n0 + tn * 16 + (lane & 15);
#pragma unroll
  for (int r = 0; r < 8; ++r) {
    int mrow = m0 + tm * 16 + r + ((lane >> 4) << 3);
    int bidx = mrow >> 10, t = mrow & 1023;
    float v = acc[r];
    if (!isGate) {
      int part = ncol >> 10;             // 0=q 1=k 2=v
      int hh = (ncol & 1023) >> 6;
      int dd = ncol & 63;
      size_t idx = (((size_t)bidx * 16 + hh) * 1024 + t) * 64 + dd;  // (B,H,T,D)
      if (part == 0)      qb[idx] = f2bf(v > 0.f ? v + 1.f : __expf(v));  // elu+1
      else if (part == 1) kb[idx] = f2bf(v > 0.f ? v + 1.f : __expf(v));
      else                vb[idx] = f2bf(v);
    } else {
      gate[((size_t)bidx * 1024 + t) * 1024 + (ncol - 3072)] = 1.f / (1.f + __expf(-v));
    }
  }
}

// ---------------------------------------------------------------------------
// Kernel 2: chunked causal linear-attention scan. One block per (b,h),
// 4 waves, chunk=64. State S (64x64) lives in WMMA f32 accumulators.
//   P = causal(Q K^T);  O = P V + Q S;  denom = rowsum(P) + q . ksum;
//   S += V^T K;  ksum += colsum(K);  out = (O / denom) * gate  -> bf16
// Q/K/V staged with global_load_async_to_lds_b128 (V lands in Ps scratch,
// then LDS->LDS transpose builds Kt/Vt before Ps is reused for scores).
// ---------------------------------------------------------------------------
__global__ __launch_bounds__(128) void k_scan(
    const __bf16* __restrict__ qb, const __bf16* __restrict__ kb,
    const __bf16* __restrict__ vb, const float* __restrict__ gate,
    __bf16* __restrict__ attn) {
  __shared__ alignas(16) __bf16 Qs[64][64];  // (t, d)
  __shared__ alignas(16) __bf16 Ks[64][64];  // (s, d)  B-operand for P
  __shared__ alignas(16) __bf16 Kt[64][64];  // (j, s)  B-operand for state upd
  __shared__ alignas(16) __bf16 Vt[64][64];  // (d, s)  A/B operand (V^T)
  __shared__ alignas(16) __bf16 Ps[64][64];  // V stage, then (t, s) scores
  __shared__ alignas(16) __bf16 Ss[64][64];  // (i, j) state, bf16 mirror
  __shared__ float ksum[64], denom[64];
  const int tid  = threadIdx.x;
  const int lane = tid & 31;
  const int tm   = tid >> 5;                   // wave = 16-row strip
  const int bidx = blockIdx.x >> 4;
  const int h    = blockIdx.x & 15;
  const size_t head_base = ((size_t)bidx * 16 + h) * 1024 * 64;

  const unsigned qlds = (unsigned)(uintptr_t)&Qs[0][0];
  const unsigned klds = (unsigned)(uintptr_t)&Ks[0][0];
  const unsigned plds = (unsigned)(uintptr_t)&Ps[0][0];

  for (int i = tid; i < 64 * 64; i += 128) Ss[i >> 6][i & 63] = f2bf(0.f);
  if (tid < 64) ksum[tid] = 0.f;
  v8f sacc[4] = {};
  __syncthreads();

  for (int c = 0; c < 16; ++c) {
    const size_t cb = head_base + (size_t)c * 64 * 64;
    // 1) async-stage Q, K (row-major) and V (into Ps scratch): 8KB each
    const unsigned long long qg = (unsigned long long)(uintptr_t)(qb + cb);
    const unsigned long long kg = (unsigned long long)(uintptr_t)(kb + cb);
    const unsigned long long vg = (unsigned long long)(uintptr_t)(vb + cb);
#pragma unroll
    for (int t = 0; t < 4; ++t) {
      const unsigned off = (unsigned)(tid + t * 128) * 16u;
      async_b128(qlds + off, qg + off);
      async_b128(klds + off, kg + off);
      async_b128(plds + off, vg + off);
    }
    wait_async0();
    __syncthreads();

    // 2) LDS->LDS transposes: Kt[j][s] = Ks[s][j], Vt[d][s] = Vstage[s][d]
    for (int i = tid * 4; i < 64 * 64; i += 128 * 4) {
      int s = i >> 6, d = i & 63;
      v4bf kr = *reinterpret_cast<const v4bf*>(&Ks[s][d]);
      v4bf vr = *reinterpret_cast<const v4bf*>(&Ps[s][d]);
#pragma unroll
      for (int j = 0; j < 4; ++j) { Kt[d + j][s] = kr[j]; Vt[d + j][s] = vr[j]; }
    }
    __syncthreads();

    // 3) P = Q @ K^T (contract over d)
    v8f pacc[4] = {};
#pragma unroll
    for (int k0 = 0; k0 < 64; k0 += 32) {
      v16bf aq = frag_ld(&Qs[tm * 16 + (lane & 15)][0], k0, lane);
#pragma unroll
      for (int tn = 0; tn < 4; ++tn) {
        v16bf bk = frag_ld(&Ks[tn * 16 + (lane & 15)][0], k0, lane);  // B(d,s)=K[s][d]
        pacc[tn] = WMMA_BF16(aq, bk, pacc[tn]);
      }
    }
    // causal mask; spill scores into Ps (V stage already consumed)
#pragma unroll
    for (int tn = 0; tn < 4; ++tn)
#pragma unroll
      for (int r = 0; r < 8; ++r) {
        int t = tm * 16 + r + ((lane >> 4) << 3);
        int s = tn * 16 + (lane & 15);
        Ps[t][s] = f2bf(s <= t ? pacc[tn][r] : 0.f);
      }
    __syncthreads();

    if (tid < 64) {  // denominator: rowsum(P) + q . running ksum
      float rs = 0.f, dq = 0.f;
      for (int s = 0; s < 64; ++s) rs += bf2f(Ps[tid][s]);
      for (int j = 0; j < 64; ++j) dq += bf2f(Qs[tid][j]) * ksum[j];
      denom[tid] = rs + dq + 1e-6f;
    }

    // 4) O = P@V + Q@S ; state update sacc += V^T @ K
    v8f oacc[4] = {};
#pragma unroll
    for (int k0 = 0; k0 < 64; k0 += 32) {
      v16bf ap  = frag_ld(&Ps[tm * 16 + (lane & 15)][0], k0, lane);
      v16bf aq2 = frag_ld(&Qs[tm * 16 + (lane & 15)][0], k0, lane);
      v16bf av  = frag_ld(&Vt[tm * 16 + (lane & 15)][0], k0, lane);  // A(i,s)=V[s][i]
#pragma unroll
      for (int tn = 0; tn < 4; ++tn) {
        v16bf bv  = frag_ld(&Vt[tn * 16 + (lane & 15)][0], k0, lane);  // B(s,d)=V[s][d]
        v16bf bs  = frag_ld(&Ss[tn * 16 + (lane & 15)][0], k0, lane);  // B(j,i)=S[i][j]
        v16bf bk2 = frag_ld(&Kt[tn * 16 + (lane & 15)][0], k0, lane);  // B(s,j)=K[s][j]
        oacc[tn] = WMMA_BF16(ap, bv, oacc[tn]);
        oacc[tn] = WMMA_BF16(aq2, bs, oacc[tn]);
        sacc[tn] = WMMA_BF16(av, bk2, sacc[tn]);
      }
    }
    __syncthreads();  // denom visible; all reads of Ss done

    // 5) gated, normalized output; refresh Ss; running ksum
    const int tg0 = c * 64;
#pragma unroll
    for (int tn = 0; tn < 4; ++tn)
#pragma unroll
      for (int r = 0; r < 8; ++r) {
        int t = tm * 16 + r + ((lane >> 4) << 3);
        int d = tn * 16 + (lane & 15);
        size_t oi = ((size_t)bidx * 1024 + (tg0 + t)) * 1024 + h * 64 + d;
        attn[oi] = f2bf((oacc[tn][r] / denom[t]) * gate[oi]);
        Ss[t][d] = f2bf(sacc[tn][r]);
      }
    if (tid < 64) {
      float s = ksum[tid];
      for (int r = 0; r < 64; ++r) s += bf2f(Kt[tid][r]);  // colsum(K), contiguous
      ksum[tid] = s;
    }
    __syncthreads();
  }
}

// ---------------------------------------------------------------------------
// Kernel 3: out = attn_bf16 @ w_out -> f32. M=2048, N=1024, K=1024.
// ---------------------------------------------------------------------------
__global__ __launch_bounds__(512) void k_out(
    const __bf16* __restrict__ attn, const float* __restrict__ w_out,
    float* __restrict__ out) {
  __shared__ alignas(16) __bf16 As[64][32];
  __shared__ alignas(16) __bf16 Bt[64][32];
  const int tid = threadIdx.x, lane = tid & 31, wave = tid >> 5;
  const int tm = wave >> 2, tn = wave & 3;
  const int m0 = (int)(blockIdx.x >> 4) * 64;
  const int n0 = (int)(blockIdx.x & 15) * 64;
  const int arow = tid >> 3, ac = (tid & 7) << 2;
  const int brow = tid >> 4, bc = (tid & 15) << 2;
  v8f acc = {};
  for (int k0 = 0; k0 < 1024; k0 += 32) {
    { // A already bf16: vector copy
      v4bf v = *reinterpret_cast<const v4bf*>(attn + (size_t)(m0 + arow) * 1024 + k0 + ac);
      *reinterpret_cast<v4bf*>(&As[arow][ac]) = v;
    }
    { // B: f32 -> bf16, transposed stage
      float4 v = *reinterpret_cast<const float4*>(w_out + (size_t)(k0 + brow) * 1024 + n0 + bc);
      Bt[bc + 0][brow] = f2bf(v.x); Bt[bc + 1][brow] = f2bf(v.y);
      Bt[bc + 2][brow] = f2bf(v.z); Bt[bc + 3][brow] = f2bf(v.w);
    }
    if (k0 + 32 < 1024) {
      __builtin_prefetch(attn + (size_t)(m0 + arow) * 1024 + k0 + 32 + ac, 0, 0);
      __builtin_prefetch(w_out + (size_t)(k0 + 32 + brow) * 1024 + n0 + bc, 0, 0);
    }
    __syncthreads();
    v16bf af = frag_ld(&As[tm * 16 + (lane & 15)][0], 0, lane);
    v16bf bf = frag_ld(&Bt[tn * 16 + (lane & 15)][0], 0, lane);
    acc = WMMA_BF16(af, bf, acc);
    __syncthreads();
  }
#pragma unroll
  for (int r = 0; r < 8; ++r) {
    int m = m0 + tm * 16 + r + ((lane >> 4) << 3);
    int n = n0 + tn * 16 + (lane & 15);
    out[(size_t)m * 1024 + n] = acc[r];
  }
}

// ---------------------------------------------------------------------------
extern "C" void kernel_launch(void* const* d_in, const int* in_sizes, int n_in,
                              void* d_out, int out_size, void* d_ws, size_t ws_size,
                              hipStream_t stream) {
  (void)in_sizes; (void)n_in; (void)out_size; (void)ws_size;
  const float* x      = (const float*)d_in[0];
  const float* w_qkv  = (const float*)d_in[1];
  const float* w_gate = (const float*)d_in[2];
  const float* w_out  = (const float*)d_in[3];
  float* out = (float*)d_out;

  // workspace layout (24 MB total):
  //  qb/kb/vb: (B,H,T,D) bf16, 4 MB each; attn: (B,T,HD) bf16 4 MB; gate f32 8 MB
  char* ws = (char*)d_ws;
  __bf16* qb   = (__bf16*)(ws);
  __bf16* kb   = (__bf16*)(ws + (size_t)(4u << 20));
  __bf16* vb   = (__bf16*)(ws + (size_t)(8u << 20));
  __bf16* attn = (__bf16*)(ws + (size_t)(12u << 20));
  float*  gate = (float*)(ws + (size_t)(16u << 20));

  k_qkv_gate<<<dim3(2048), dim3(512), 0, stream>>>(x, w_qkv, w_gate, qb, kb, vb, gate);
  k_scan<<<dim3(32), dim3(128), 0, stream>>>(qb, kb, vb, gate, attn);
  k_out<<<dim3(512), dim3(512), 0, stream>>>(attn, w_out, out);
}
